// SudokuSolver_77988016161256
// MI455X (gfx1250) — matrix operand order repaired and checked by reading, not compiled, not run on hardware
//
#include <hip/hip_runtime.h>

// ---------------------------------------------------------------------------
// Sudoku iterative-MLP solver for MI455X (gfx1250), wave32 + WMMA bf16.
//
//  * one workgroup (256 threads = 8 wave32) per board; all 4 solver
//    iterations fused in-kernel (boards are fully independent).
//  * weights converted fp32->bf16 once per launch and pre-swizzled into the
//    V_WMMA_F32_16X16X32_BF16 B-fragment lane layout (coalesced 32B/lane).
//  * hidden-layer GEMM is 6x1 register blocked: each wave holds 6
//    accumulators (all M-tiles of one N-tile), so each B fragment is loaded
//    once per 6 WMMAs and consecutive WMMAs are independent (full XDL rate).
//  * B stream double-buffered in registers to overlap L2 latency with WMMA.
//  * activations (96x512 bf16) ping-pong in LDS; A-frags = 2x ds_load_b128.
//  * occupancy is LDS-bound (one 215KB block per 320KB WGP), so
//    __launch_bounds__(256, 1) grants max VGPRs and eliminates spills.
// ---------------------------------------------------------------------------

typedef __bf16  v16bf __attribute__((ext_vector_type(16)));
typedef __bf16  v8bf  __attribute__((ext_vector_type(8)));
typedef float   v8f   __attribute__((ext_vector_type(8)));

#define NQ     81          // cells per board
#define ND     9           // digits
#define H      512         // hidden width
#define MPAD   96          // 81 rows padded to 6 M-tiles
#define NMT    6           // M tiles
#define NNT    32          // N tiles for H=512
#define NKT    16          // K tiles for K=512 (K=32 per WMMA)
#define BLK    512         // bf16 elems per packed 16x16x32 weight block
#define NBOARD 512
#define XPERB  (NQ * ND)   // 729
#define XTOTAL (NBOARD * XPERB)

// packed weight offsets inside d_ws (bf16 element units)
#define OFF_W1   ((size_t)0)                      // K=27->32,  N=512 : 1*32  blocks
#define SZ_W1    ((size_t)(1 * 32 * BLK))         // 16384
#define SZ_WH    ((size_t)(16 * 32 * BLK))        // 262144 per hidden layer
#define OFF_WH(i) (SZ_W1 + (size_t)(i) * SZ_WH)
#define OFF_W10  (SZ_W1 + (size_t)8 * SZ_WH)      // K=512, N=9->16 : 16*1 blocks

// ---------------------------------------------------------------------------
__device__ __forceinline__ v8f wmma_bf16(v16bf a, v16bf b, v8f c) {
  return __builtin_amdgcn_wmma_f32_16x16x32_bf16(
      /*neg_a=*/false, a, /*neg_b=*/false, b,
      /*c_mod=*/(short)0, c, /*reuse_a=*/false, /*reuse_b=*/false);
}

// A fragment (16x32 bf16, row-major activations in LDS, row stride `stride`).
// Lane group g (= lane>>4) holds K in {8g..8g+7} U {16+8g..16+8g+7}: two
// contiguous 16B runs -> two ds_load_b128.
__device__ __forceinline__ v16bf load_a(const __bf16* act, int stride,
                                        int mtile, int kt, int lane) {
  const int g = lane >> 4;
  const int m = mtile * 16 + (lane & 15);
  const __bf16* p = act + (size_t)m * stride + kt * 32 + g * 8;
  v8bf lo = *(const v8bf*)p;
  v8bf hi = *(const v8bf*)(p + 16);
  v16bf a;
#pragma unroll
  for (int i = 0; i < 8; ++i) { a[i] = lo[i]; a[i + 8] = hi[i]; }
  return a;
}

// ---------------------------------------------------------------------------
// Weight pre-pack: fp32 (Kreal x Nreal row-major) -> bf16 WMMA B-fragment
// blocks. Block (ntile,ktile) stores, at [lane*16 + e]:
//   k = ktile*32 + 8*(lane>>4) + (e&7) + 16*(e>>3),  n = ntile*16 + (lane&15)
// ---------------------------------------------------------------------------
__global__ void pack_weights_kernel(const float* __restrict__ w,
                                    __bf16* __restrict__ out,
                                    int Kreal, int Nreal, int nKt, int nNt) {
  int idx = blockIdx.x * blockDim.x + threadIdx.x;
  int total = nKt * nNt * BLK;
  if (idx >= total) return;
  int blk  = idx >> 9;          // /512
  int le   = idx & 511;
  int lane = le >> 4;
  int e    = le & 15;
  int ntile = blk / nKt;
  int ktile = blk % nKt;
  int k = ktile * 32 + 8 * (lane >> 4) + (e & 7) + 16 * (e >> 3);
  int n = ntile * 16 + (lane & 15);
  float v = (k < Kreal && n < Nreal) ? w[(size_t)k * Nreal + n] : 0.0f;
  out[idx] = (__bf16)v;
}

// ---------------------------------------------------------------------------
// Hidden layer: out = relu(in @ W + b).  Wave w owns N-tiles {w, w+8, ...};
// for each N-tile it keeps all 6 M-tile accumulators live, B double-buffered.
// ---------------------------------------------------------------------------
__device__ __forceinline__ void layer_hidden(const __bf16* __restrict__ in,
                                             __bf16* __restrict__ out,
                                             const __bf16* __restrict__ wpack,
                                             const float* __restrict__ bias,
                                             int wave, int lane) {
  const int g  = lane >> 4;
  const int nl = lane & 15;
  const v8f zero = {0.f, 0.f, 0.f, 0.f, 0.f, 0.f, 0.f, 0.f};

  for (int nt = wave; nt < NNT; nt += 8) {
    const __bf16* wb = wpack + (size_t)nt * (NKT * BLK) + lane * 16;
    v8f c[NMT];
#pragma unroll
    for (int m = 0; m < NMT; ++m) c[m] = zero;

    v16bf b = *(const v16bf*)wb;              // prime the B pipeline
#pragma unroll 2
    for (int kt = 0; kt < NKT - 1; ++kt) {
      __builtin_prefetch(wb + (size_t)(kt + 2) * BLK, 0, 0);
      v16bf bn = *(const v16bf*)(wb + (size_t)(kt + 1) * BLK);  // next B
#pragma unroll
      for (int m = 0; m < NMT; ++m) {         // 6 independent WMMAs / B-load
        v16bf a = load_a(in, H, m, kt, lane);
        c[m] = wmma_bf16(a, b, c[m]);
      }
      b = bn;
    }
#pragma unroll
    for (int m = 0; m < NMT; ++m) {           // epilogue K-step
      v16bf a = load_a(in, H, m, NKT - 1, lane);
      c[m] = wmma_bf16(a, b, c[m]);
    }

    const int n  = nt * 16 + nl;
    const float bv = bias[n];
#pragma unroll
    for (int m = 0; m < NMT; ++m) {
#pragma unroll
      for (int r = 0; r < 8; ++r) {
        float v = c[m][r] + bv;
        out[(size_t)(m * 16 + g * 8 + r) * H + n] = (__bf16)(v > 0.f ? v : 0.f);
      }
    }
  }
}

// ---------------------------------------------------------------------------
// Main fused solver kernel: one block per board.
// ---------------------------------------------------------------------------
__global__ void __launch_bounds__(256, 1)
sudoku_solver_kernel(const float* __restrict__ x_in,
                     const int* __restrict__ n_iters_p,
                     const __bf16* __restrict__ wpack,
                     const float* __restrict__ b1,
                     const float* __restrict__ b2,  const float* __restrict__ b3,
                     const float* __restrict__ b4,  const float* __restrict__ b5,
                     const float* __restrict__ b6,  const float* __restrict__ b7,
                     const float* __restrict__ b8,  const float* __restrict__ b9,
                     const float* __restrict__ b10,
                     float* __restrict__ out) {
  __shared__ __bf16 sActA[MPAD * H];     // 96 KB
  __shared__ __bf16 sActB[MPAD * H];     // 96 KB
  __shared__ __bf16 sF[MPAD * 32];       // padded 81x27 features
  __shared__ float  sLogits[MPAD * 16];  // padded 81x9 logits
  __shared__ float  sX[XPERB];           // board state
  __shared__ float  sXp[XPERB];          // x_pred
  __shared__ float  sVal[NQ];
  __shared__ int    sPos[NQ];

  const int tid   = threadIdx.x;
  const int lane  = tid & 31;
  const int wave  = tid >> 5;
  const int board = blockIdx.x;
  const int iters = *n_iters_p;
  const v8f zero  = {0.f, 0.f, 0.f, 0.f, 0.f, 0.f, 0.f, 0.f};

  const float* hb[8] = {b2, b3, b4, b5, b6, b7, b8, b9};

  // ---- init board state, x_pred, and zero the padded feature buffer ----
  for (int i = tid; i < XPERB; i += 256) {
    float v = x_in[(size_t)board * XPERB + i];
    sX[i]  = v;
    sXp[i] = v;
  }
  for (int i = tid; i < MPAD * 32; i += 256) sF[i] = (__bf16)0.0f;
  __syncthreads();

  for (int it = 0; it < iters; ++it) {
    // ---- constraint contraction: f[q, t*9+n] = sum over 9 peer cells ----
    for (int i = tid; i < NQ * 27; i += 256) {
      const int q   = i / 27;
      const int rem = i % 27;
      const int t   = rem / 9;
      const int n   = rem % 9;
      const int r   = q / 9, cc = q % 9;
      float s = 0.0f;
#pragma unroll
      for (int j = 0; j < 9; ++j) {
        int cell;
        if (t == 0)       cell = r * 9 + j;                                        // row
        else if (t == 1)  cell = j * 9 + cc;                                       // col
        else              cell = ((r / 3) * 3 + j / 3) * 9 + (cc / 3) * 3 + j % 3; // box
        s += sX[cell * ND + n];
      }
      sF[q * 32 + rem] = (__bf16)s;
    }
    __syncthreads();

    // ---- layer 1: (96x32) @ (32x512) + b1, relu -> sActA (B reused 6x) ----
    {
      const int g = lane >> 4, nl = lane & 15;
      for (int nt = wave; nt < NNT; nt += 8) {
        v16bf b = *(const v16bf*)(wpack + OFF_W1 + (size_t)nt * BLK + lane * 16);
        const int n = nt * 16 + nl;
        const float bv = b1[n];
#pragma unroll
        for (int m = 0; m < NMT; ++m) {
          v16bf a = load_a(sF, 32, m, 0, lane);
          v8f c = wmma_bf16(a, b, zero);
#pragma unroll
          for (int r = 0; r < 8; ++r) {
            float v = c[r] + bv;
            sActA[(size_t)(m * 16 + g * 8 + r) * H + n] = (__bf16)(v > 0.f ? v : 0.f);
          }
        }
      }
    }
    __syncthreads();

    // ---- 8 hidden layers, ping-pong sActA <-> sActB ----
    const __bf16* cur = sActA;
    __bf16*       nxt = sActB;
    for (int l = 0; l < 8; ++l) {
      layer_hidden(cur, nxt, wpack + OFF_WH(l), hb[l], wave, lane);
      __syncthreads();
      const __bf16* tmp = cur; cur = nxt; nxt = (__bf16*)tmp;
    }

    // ---- layer 10: (96x512) @ (512x16pad) + b10 -> fp32 logits ----
    if (wave < NMT) {
      const int mtile = wave;
      const __bf16* wb = wpack + OFF_W10 + lane * 16;
      v8f c = zero;
      v16bf b = *(const v16bf*)wb;
#pragma unroll 4
      for (int kt = 0; kt < NKT - 1; ++kt) {
        v16bf bn = *(const v16bf*)(wb + (size_t)(kt + 1) * BLK);
        v16bf a  = load_a(cur, H, mtile, kt, lane);
        c = wmma_bf16(a, b, c);
        b = bn;
      }
      {
        v16bf a = load_a(cur, H, mtile, NKT - 1, lane);
        c = wmma_bf16(a, b, c);
      }
      const int g = lane >> 4;
      const int n = lane & 15;
      const float bv = (n < ND) ? b10[n] : 0.0f;
#pragma unroll
      for (int r = 0; r < 8; ++r)
        sLogits[(mtile * 16 + g * 8 + r) * 16 + n] = c[r] + bv;
    }
    __syncthreads();

    // ---- softmax + per-cell confidence (threads 0..80) ----
    if (tid < NQ) {
      const int q = tid;
      float es = 0.0f;
#pragma unroll
      for (int n = 0; n < ND; ++n) es += sX[q * ND + n];
      const bool empty = (es == 0.0f);

      float mx = sLogits[q * 16];
#pragma unroll
      for (int n = 1; n < ND; ++n) mx = fmaxf(mx, sLogits[q * 16 + n]);
      float e[ND], den = 0.0f;
#pragma unroll
      for (int n = 0; n < ND; ++n) { e[n] = __expf(sLogits[q * 16 + n] - mx); den += e[n]; }
      const float inv = 1.0f / den;

      float best = -1.0f; int bi = 0;
#pragma unroll
      for (int n = 0; n < ND; ++n) {
        const float s = e[n] * inv;
        if (empty) sXp[q * ND + n] = s;      // x_pred = where(empty, softmax, x_pred)
        if (s > best) { best = s; bi = n; }
      }
      sVal[q] = empty ? best : 0.0f;          // s = where(empty, softmax, 0)
      sPos[q] = bi;
    }
    __syncthreads();

    // ---- pick most confident cell across board; commit a one-hot 1 ----
    if (tid == 0) {
      float best = sVal[0]; int bq = 0;
      for (int q = 1; q < NQ; ++q)
        if (sVal[q] > best) { best = sVal[q]; bq = q; }   // first-max tie-break
      if (best > 0.0f)                                     // has_empty
        sX[bq * ND + sPos[bq]] = 1.0f;
    }
    __syncthreads();
  }

  // ---- write outputs: [x_pred | x] concatenated ----
  for (int i = tid; i < XPERB; i += 256) {
    out[(size_t)board * XPERB + i]          = sXp[i];
    out[XTOTAL + (size_t)board * XPERB + i] = sX[i];
  }
}

// ---------------------------------------------------------------------------
// Launch: pack weights into d_ws, then run one block per board.
// ---------------------------------------------------------------------------
extern "C" void kernel_launch(void* const* d_in, const int* in_sizes, int n_in,
                              void* d_out, int out_size, void* d_ws, size_t ws_size,
                              hipStream_t stream) {
  (void)in_sizes; (void)n_in; (void)out_size; (void)ws_size;
  const float* x       = (const float*)d_in[0];
  // d_in[1] = cm (recomputed analytically in-kernel)
  const int*   n_iters = (const int*)d_in[22];
  __bf16*      wp      = (__bf16*)d_ws;

  auto pack = [&](const float* w, size_t off, int K, int N, int nKt, int nNt) {
    int total = nKt * nNt * BLK;
    pack_weights_kernel<<<(total + 255) / 256, 256, 0, stream>>>(w, wp + off, K, N, nKt, nNt);
  };

  pack((const float*)d_in[2],  OFF_W1,  27,  H, 1,  NNT);          // w1
  for (int i = 0; i < 8; ++i)                                      // w2..w9
    pack((const float*)d_in[4 + 2 * i], OFF_WH(i), H, H, NKT, NNT);
  pack((const float*)d_in[20], OFF_W10, H, ND, NKT, 1);            // w10

  sudoku_solver_kernel<<<NBOARD, 256, 0, stream>>>(
      x, n_iters, wp,
      (const float*)d_in[3],  (const float*)d_in[5],  (const float*)d_in[7],
      (const float*)d_in[9],  (const float*)d_in[11], (const float*)d_in[13],
      (const float*)d_in[15], (const float*)d_in[17], (const float*)d_in[19],
      (const float*)d_in[21],
      (float*)d_out);
}
